// NMT_4715874091847
// MI455X (gfx1250) — matrix operand order repaired and checked by reading, not compiled, not run on hardware
//
#include <hip/hip_runtime.h>
#include <hip/hip_bf16.h>
#include <math.h>

typedef __attribute__((ext_vector_type(16))) __bf16 v16bf;
typedef __attribute__((ext_vector_type(8)))  __bf16 v8bf;
typedef __attribute__((ext_vector_type(8)))  float  v8f;

#define LL 512           // sequence length
#define HH 512           // hidden (2h)
#define BLK_M 32         // rows of A per workgroup
#define KC 32            // WMMA K chunk (bf16 16x16x32)
#define SROW 516         // S LDS row stride (fp32, padded)
#define PROW 520         // A'/P LDS row stride (bf16, padded, mult of 8)
#define BROW 40          // B-stage LDS row stride (bf16, mult of 8)

// workspace layout (float offsets)
#define WS_ROWPROJ 0         // [24][512]  A-side projections (X @ w[:h2])
#define WS_COLPROJ 12288     // [24][512]  B-side projections (Y @ w[h2:2h2])
#define WS_ROWMAX  24576     // [24][512]  max_j S per row (for beta softmax)
#define WS_B2A     36864     // [24][512]  b2a vectors

// dynamic LDS layout (byte offsets) for the core kernel
#define OFF_S    0           // 32*516*4   = 66048
#define OFF_P    66048       // 32*520*2   = 33280  (A' for GEMM1, P for GEMM2)
#define OFF_BST  99328       // 512*40*2   = 40960
#define OFF_ROWA 140288      // 32*4
#define OFF_COLB 140416      // 512*4
#define OFF_RED  142464      // 32*8*4
#define SMEM_CORE 143488

__device__ __forceinline__ v16bf cat8(v8bf a, v8bf b) {
  return __builtin_shufflevector(a, b, 0,1,2,3,4,5,6,7,8,9,10,11,12,13,14,15);
}

// ---------------------------------------------------------------------------
// Kernel 0: per (pair,batch) projections rowA = Aside@w[:h2], colB = Bside@w[h2:2h2]
// ---------------------------------------------------------------------------
__global__ void tri_proj_kernel(const float* __restrict__ ea,
                                const float* __restrict__ eb,
                                const float* __restrict__ ec,
                                const float* __restrict__ w_ab,
                                const float* __restrict__ w_ac,
                                const float* __restrict__ w_bc,
                                float* __restrict__ ws)
{
  const int p = blockIdx.x, b = blockIdx.y;
  const float* A; const float* Bx; const float* w;
  if (p == 0)      { A = ea; Bx = eb; w = w_ab; }
  else if (p == 1) { A = ea; Bx = ec; w = w_ac; }
  else             { A = eb; Bx = ec; w = w_bc; }
  const size_t base = (size_t)b * LL * HH;
  const int tid = threadIdx.x;
  const int lane = tid & 31, wave = tid >> 5;
  for (int r = wave; r < 2 * LL; r += 8) {
    const float* src; const float* wv; float* dst;
    if (r < LL) { src = A + base + (size_t)r * HH;        wv = w;      dst = ws + WS_ROWPROJ + (b*3+p)*LL + r; }
    else        { src = Bx + base + (size_t)(r-LL) * HH;  wv = w + HH; dst = ws + WS_COLPROJ + (b*3+p)*LL + (r-LL); }
    float s = 0.f;
    #pragma unroll
    for (int q = 0; q < 4; ++q) {
      const int k = lane * 16 + q * 4;
      float4 a4 = reinterpret_cast<const float4*>(src + k)[0];
      float4 w4 = reinterpret_cast<const float4*>(wv  + k)[0];
      s += a4.x*w4.x + a4.y*w4.y + a4.z*w4.z + a4.w*w4.w;
    }
    #pragma unroll
    for (int off = 16; off > 0; off >>= 1) s += __shfl_xor(s, off, 32);
    if (lane == 0) *dst = s;
  }
}

// ---------------------------------------------------------------------------
// Kernel 1 (core): per (batch,pair,32-row block):
//   S = (A*wM) @ Bx^T + rowA + colB      (bf16 WMMA, fp32 acc)
//   P = softmax_rows(S); rowmax -> ws
//   a2b = P @ Bx  -> written straight into the output slot
// 8 waves; wave w owns tile-row tm=w&1, tile-cols [(w>>1)*8, +8) of a 2x32 tile grid.
// ---------------------------------------------------------------------------
__global__ void __launch_bounds__(256)
tri_pair_core_kernel(const float* __restrict__ ea, const float* __restrict__ eb,
                     const float* __restrict__ ec,
                     const float* __restrict__ w_ab, const float* __restrict__ w_ac,
                     const float* __restrict__ w_bc,
                     float* __restrict__ out, float* __restrict__ ws)
{
  extern __shared__ char smem[];
  float*  S    = reinterpret_cast<float*>(smem + OFF_S);
  __bf16* P    = reinterpret_cast<__bf16*>(smem + OFF_P);
  __bf16* Bst  = reinterpret_cast<__bf16*>(smem + OFF_BST);
  float*  rowA = reinterpret_cast<float*>(smem + OFF_ROWA);
  float*  colB = reinterpret_cast<float*>(smem + OFF_COLB);
  float*  red  = reinterpret_cast<float*>(smem + OFF_RED);

  const int ib = blockIdx.x, p = blockIdx.y, b = blockIdx.z;
  const int i0 = ib * BLK_M;
  const float* A; const float* Bx; const float* w; int seqbase, slot;
  if (p == 0)      { A = ea; Bx = eb; w = w_ab; seqbase = 0;   slot = 512;  }
  else if (p == 1) { A = ea; Bx = ec; w = w_ac; seqbase = 0;   slot = 1024; }
  else             { A = eb; Bx = ec; w = w_bc; seqbase = 512; slot = 1024; }
  const float* wM = w + 2 * HH;
  const size_t bbase = (size_t)b * LL * HH;
  A += bbase; Bx += bbase;

  const int tid  = threadIdx.x;
  const int lane = tid & 31, wave = tid >> 5;
  const int tm  = wave & 1;          // tile-row of this wave
  const int tnb = (wave >> 1) * 8;   // first tile-col of this wave
  const int hi  = lane >> 4;         // which 16-lane half
  const int ln  = lane & 15;

  // Stage A' = A_rowblock * diag(wM) as bf16 (WMMA A-operand friendly: row-major).
  for (int idx = tid; idx < BLK_M * (HH/4); idx += 256) {
    const int r = idx >> 7, c4 = (idx & 127) * 4;
    float4 a4 = reinterpret_cast<const float4*>(A + (size_t)(i0 + r) * HH + c4)[0];
    float4 w4 = reinterpret_cast<const float4*>(wM + c4)[0];
    __bf16* d = P + r * PROW + c4;
    d[0] = (__bf16)(a4.x * w4.x); d[1] = (__bf16)(a4.y * w4.y);
    d[2] = (__bf16)(a4.z * w4.z); d[3] = (__bf16)(a4.w * w4.w);
  }
  if (tid < BLK_M) rowA[tid] = ws[WS_ROWPROJ + (b*3+p)*LL + i0 + tid];
  for (int j = tid; j < LL; j += 256) colB[j] = ws[WS_COLPROJ + (b*3+p)*LL + j];

  v8f acc[8];
  #pragma unroll
  for (int t = 0; t < 8; ++t)
    #pragma unroll
    for (int r = 0; r < 8; ++r) acc[t][r] = 0.f;

  // -------- GEMM1: S = A' @ Bx^T --------
  for (int kk = 0; kk < HH; kk += KC) {
    // stage Bst[j][kc] = bf16(Bx[j][kk+kc])  (row = WMMA-N index, contiguous K)
    for (int idx = tid; idx < LL * (KC/4); idx += 256) {
      const int j = idx >> 3, c4 = (idx & 7) * 4;
      float4 v = reinterpret_cast<const float4*>(Bx + (size_t)j * HH + kk + c4)[0];
      __bf16* d = Bst + j * BROW + c4;
      d[0] = (__bf16)v.x; d[1] = (__bf16)v.y; d[2] = (__bf16)v.z; d[3] = (__bf16)v.w;
    }
    __syncthreads();
    // A-frag: lane = (row ln, K-half hi): K = kk+hi*16+{0..15}
    const __bf16* ap = P + (tm*16 + ln) * PROW + kk + hi*16;
    v16bf af = cat8(*reinterpret_cast<const v8bf*>(ap),
                    *reinterpret_cast<const v8bf*>(ap + 8));
    #pragma unroll
    for (int t = 0; t < 8; ++t) {
      // B-frag: lane = (col ln, K-half hi): 16 contiguous bf16
      const __bf16* bp = Bst + ((tnb + t)*16 + ln) * BROW + hi*16;
      v16bf bfv = cat8(*reinterpret_cast<const v8bf*>(bp),
                       *reinterpret_cast<const v8bf*>(bp + 8));
      acc[t] = __builtin_amdgcn_wmma_f32_16x16x32_bf16(false, af, false, bfv,
                                                       (short)0, acc[t], false, false);
    }
    __syncthreads();
  }

  // bias add + spill S to LDS (C/D layout: element (vgpr r, lane) = (M=r+8*hi, N=ln))
  #pragma unroll
  for (int t = 0; t < 8; ++t) {
    const int col = (tnb + t) * 16 + ln;
    const float cb = colB[col];
    #pragma unroll
    for (int r = 0; r < 8; ++r) {
      const int row = tm*16 + hi*8 + r;
      S[row * SROW + col] = acc[t][r] + rowA[row] + cb;
    }
  }
  __syncthreads();

  // -------- row softmax: P = softmax(S), rowmax -> ws --------
  {
    const int srow = tid >> 3, sl = tid & 7;   // 8 threads per row
    float m = -3.4e38f;
    for (int c = sl; c < LL; c += 8) m = fmaxf(m, S[srow*SROW + c]);
    red[srow*8 + sl] = m;
    __syncthreads();
    float rm = red[srow*8];
    #pragma unroll
    for (int k = 1; k < 8; ++k) rm = fmaxf(rm, red[srow*8 + k]);
    if (sl == 0) ws[WS_ROWMAX + (b*3+p)*LL + i0 + srow] = rm;
    float sum = 0.f;
    for (int c = sl; c < LL; c += 8) sum += __expf(S[srow*SROW + c] - rm);
    __syncthreads();
    red[srow*8 + sl] = sum;
    __syncthreads();
    float rs = 0.f;
    #pragma unroll
    for (int k = 0; k < 8; ++k) rs += red[srow*8 + k];
    const float inv = 1.f / rs;
    for (int c = sl; c < LL; c += 8)
      P[srow*PROW + c] = (__bf16)(__expf(S[srow*SROW + c] - rm) * inv);
  }

  #pragma unroll
  for (int t = 0; t < 8; ++t)
    #pragma unroll
    for (int r = 0; r < 8; ++r) acc[t][r] = 0.f;

  // -------- GEMM2: a2b = P @ Bx --------
  for (int kk = 0; kk < LL; kk += KC) {
    // stage transposed: Bst[d][jj] = bf16(Bx[kk+jj][d])  (row = output col d)
    for (int idx = tid; idx < KC * (HH/4); idx += 256) {
      const int jj = idx >> 7, c4 = (idx & 127) * 4;
      float4 v = reinterpret_cast<const float4*>(Bx + (size_t)(kk + jj) * HH + c4)[0];
      Bst[(c4+0)*BROW + jj] = (__bf16)v.x;
      Bst[(c4+1)*BROW + jj] = (__bf16)v.y;
      Bst[(c4+2)*BROW + jj] = (__bf16)v.z;
      Bst[(c4+3)*BROW + jj] = (__bf16)v.w;
    }
    __syncthreads();   // also orders softmax P writes vs P reads below
    const __bf16* ap = P + (tm*16 + ln) * PROW + kk + hi*16;
    v16bf af = cat8(*reinterpret_cast<const v8bf*>(ap),
                    *reinterpret_cast<const v8bf*>(ap + 8));
    #pragma unroll
    for (int t = 0; t < 8; ++t) {
      const __bf16* bp = Bst + ((tnb + t)*16 + ln) * BROW + hi*16;
      v16bf bfv = cat8(*reinterpret_cast<const v8bf*>(bp),
                       *reinterpret_cast<const v8bf*>(bp + 8));
      acc[t] = __builtin_amdgcn_wmma_f32_16x16x32_bf16(false, af, false, bfv,
                                                       (short)0, acc[t], false, false);
    }
    __syncthreads();
  }

  // write a2b tiles straight into the output slot (avoids a 25 MB ws round trip)
  {
    const size_t orow0 = (size_t)b * 1536 + seqbase + i0;
    #pragma unroll
    for (int t = 0; t < 8; ++t) {
      const int col = (tnb + t) * 16 + ln;
      #pragma unroll
      for (int r = 0; r < 8; ++r) {
        const int row = tm*16 + hi*8 + r;
        out[(orow0 + row) * 2560 + slot + col] = acc[t][r];
      }
    }
  }
}

// ---------------------------------------------------------------------------
// Kernel 2: beta = softmax_i(rowmax); b2a[d] = sum_i beta[i]*Aside[i][d]
// ---------------------------------------------------------------------------
__global__ void tri_b2a_kernel(const float* __restrict__ ea,
                               const float* __restrict__ eb,
                               const float* __restrict__ ec,
                               float* __restrict__ ws)
{
  const int p = blockIdx.x, b = blockIdx.y;
  __shared__ float beta[LL];
  __shared__ float sred[8];
  const int tid = threadIdx.x, lane = tid & 31, wave = tid >> 5;
  const float* m = ws + WS_ROWMAX + (b*3+p)*LL;
  const float m0 = m[tid], m1 = m[tid + 256];
  float v = fmaxf(m0, m1);
  #pragma unroll
  for (int off = 16; off > 0; off >>= 1) v = fmaxf(v, __shfl_xor(v, off, 32));
  if (lane == 0) sred[wave] = v;
  __syncthreads();
  float M = sred[0];
  #pragma unroll
  for (int k = 1; k < 8; ++k) M = fmaxf(M, sred[k]);
  const float e0 = __expf(m0 - M), e1 = __expf(m1 - M);
  float s = e0 + e1;
  __syncthreads();
  #pragma unroll
  for (int off = 16; off > 0; off >>= 1) s += __shfl_xor(s, off, 32);
  if (lane == 0) sred[wave] = s;
  __syncthreads();
  float Ssum = 0.f;
  #pragma unroll
  for (int k = 0; k < 8; ++k) Ssum += sred[k];
  const float inv = 1.f / Ssum;
  beta[tid] = e0 * inv; beta[tid + 256] = e1 * inv;
  __syncthreads();
  const float* A = ((p == 2) ? eb : ea) + (size_t)b * LL * HH;
  for (int d = tid; d < HH; d += 256) {
    float acc = 0.f;
    for (int i = 0; i < LL; ++i) acc += beta[i] * A[(size_t)i * HH + d];
    ws[WS_B2A + (b*3+p)*LL + d] = acc;
  }
}

// ---------------------------------------------------------------------------
// Kernel 3: assemble [X, first, second, X*first, X*second] per row
// ---------------------------------------------------------------------------
__global__ void tri_assemble_kernel(const float* __restrict__ ea,
                                    const float* __restrict__ eb,
                                    const float* __restrict__ ec,
                                    const float* __restrict__ ws,
                                    float* __restrict__ out)
{
  const int row = blockIdx.x, b = blockIdx.y, tid = threadIdx.x;
  const int seq = row >> 9, i = row & 511;
  const float* X = (seq == 0 ? ea : (seq == 1 ? eb : ec)) + ((size_t)b * LL + i) * HH;
  float* orow = out + ((size_t)b * 1536 + row) * 2560;
  const float* f1; const float* f2; bool bc1, bc2;
  if (seq == 0)      { f1 = orow + 512;                 bc1 = false; f2 = orow + 1024;               bc2 = false; }
  else if (seq == 1) { f1 = ws + WS_B2A + (b*3+0)*LL;   bc1 = true;  f2 = orow + 1024;               bc2 = false; }
  else               { f1 = ws + WS_B2A + (b*3+1)*LL;   bc1 = true;  f2 = ws + WS_B2A + (b*3+2)*LL;  bc2 = true;  }
  for (int d = tid; d < HH; d += 256) {
    const float x  = X[d];
    const float v1 = f1[d];
    const float v2 = f2[d];
    orow[d] = x;
    if (bc1) orow[512  + d] = v1;
    if (bc2) orow[1024 + d] = v2;
    orow[1536 + d] = x * v1;
    orow[2048 + d] = x * v2;
  }
}

// ---------------------------------------------------------------------------
extern "C" void kernel_launch(void* const* d_in, const int* in_sizes, int n_in,
                              void* d_out, int out_size, void* d_ws, size_t ws_size,
                              hipStream_t stream)
{
  (void)in_sizes; (void)n_in; (void)out_size; (void)ws_size;
  const float* ea   = (const float*)d_in[0];
  const float* eb   = (const float*)d_in[1];
  const float* ec   = (const float*)d_in[2];
  const float* w_ab = (const float*)d_in[3];
  const float* w_ac = (const float*)d_in[4];
  const float* w_bc = (const float*)d_in[5];
  float* out = (float*)d_out;
  float* ws  = (float*)d_ws;   // needs 49152 floats = 192 KB

  hipFuncSetAttribute(reinterpret_cast<const void*>(tri_pair_core_kernel),
                      hipFuncAttributeMaxDynamicSharedMemorySize, SMEM_CORE);

  tri_proj_kernel<<<dim3(3, 8), 256, 0, stream>>>(ea, eb, ec, w_ab, w_ac, w_bc, ws);
  tri_pair_core_kernel<<<dim3(LL / BLK_M, 3, 8), 256, SMEM_CORE, stream>>>(
      ea, eb, ec, w_ab, w_ac, w_bc, out, ws);
  tri_b2a_kernel<<<dim3(3, 8), 256, 0, stream>>>(ea, eb, ec, ws);
  tri_assemble_kernel<<<dim3(1536, 8), 256, 0, stream>>>(ea, eb, ec, ws, out);
}
